// SLSTM_28363964023437
// MI455X (gfx1250) — matrix operand order repaired and checked by reading, not compiled, not run on hardware
//
#include <hip/hip_runtime.h>
#include <hip/hip_bf16.h>

typedef __attribute__((ext_vector_type(8)))  float        v8f;
typedef __attribute__((ext_vector_type(8)))  unsigned int v8u;
typedef __attribute__((ext_vector_type(16))) __bf16       v16bf;
typedef unsigned short u16;
typedef unsigned int   u32;

#define T_STEPS 128
#define BATCH   256
#define HID     1024
#define OUTD    10
#define G4H     4096
#define KT32    32   // HID / 32  (K tiles per 1024-deep GEMM)

// ---------- helpers ----------

__device__ __forceinline__ u16 f2bf(float f) {
  u32 u = __float_as_uint(f);
  u32 r = u + 0x7FFFu + ((u >> 16) & 1u);   // round-to-nearest-even
  return (u16)(r >> 16);
}

__device__ __forceinline__ float sigmoidf_(float x) {
  return 1.0f / (1.0f + __expf(-x));
}

// WMMA A-fragment (16M x 32K, bf16) packed layout: halfword index for element (b,h)
// lane<16 holds K={0..7,16..23}; lane>=16 holds K={8..15,24..31}; 2 bf16 per dword.
__device__ __forceinline__ u32 apack_idx(int b, int h) {
  int mt  = b >> 4;
  int kt  = h >> 5;
  int kin = h & 31;
  int lane = (b & 15) + ((kin & 8) ? 16 : 0);
  int j    = ((kin & 7) >> 1) + ((kin >> 4) << 2);
  int hi   = kin & 1;
  u32 dw = ((u32)(mt * KT32 + kt) * 32u + (u32)lane) * 8u + (u32)j;
  return dw * 2u + (u32)hi;
}

// ---------- weight packing: row-major W[N,K] (C[b,n]=sum_k A[b,k]*W[n,k]) ----------
// WMMA B-fragment (32K x 16N): lane<16 K={0..7,16..23}; lane>=16 K={8..15,24..31}; 2 bf16 per dword.
__global__ void pack_w_kernel(const float* __restrict__ W, u16* __restrict__ out, int NK) {
  int idx = blockIdx.x * blockDim.x + threadIdx.x;
  if (idx >= NK) return;
  int n = idx >> 10;          // K == 1024 always
  int k = idx & 1023;
  int nt  = n >> 4;
  int kt  = k >> 5;
  int kin = k & 31;
  int lane = (n & 15) + ((kin & 8) ? 16 : 0);
  int j    = ((kin & 7) >> 1) + ((kin >> 4) << 2);
  int hi   = kin & 1;
  u32 dw = ((u32)(nt * KT32 + kt) * 32u + (u32)lane) * 8u + (u32)j;
  out[dw * 2u + (u32)hi] = f2bf(W[idx]);
}

// ---------- async staging of one packed B tile (1 KB) into LDS ----------
// Two b128 async issues per lane cover 32 lanes * 16B * 2 = 1024 bytes (linear copy).
__device__ __forceinline__ void stage_tile(const u32* __restrict__ Wp, int nt, int kk,
                                           u32* lds_dst, int lane) {
  const u32* s0 = Wp + ((u32)(nt * KT32 + kk) * 256u) + (u32)lane * 4u;
  u32 l0 = (u32)(size_t)lds_dst + (u32)lane * 16u;
  asm volatile("global_load_async_to_lds_b128 %0, %1, off"
               :: "v"(l0), "v"(s0) : "memory");
  asm volatile("global_load_async_to_lds_b128 %0, %1, off"
               :: "v"(l0 + 512u), "v"(s0 + 128) : "memory");
}

// ---------- bf16 WMMA GEMM: C[B, N] = A1*W1^T (+ A2*W2^T) + bias1 (+ bias2) ----------
// Block = 8 waves = all 256 batch rows (wave w -> rows 32w..32w+31) for one 64-col strip.
// B tiles for each k-step are async-staged into LDS once per block (double-buffered)
// so each weight byte is read from L2 exactly once per GEMM.
__global__ void __launch_bounds__(256)
gemm_bf16_kernel(const u32* __restrict__ Ap1, const u32* __restrict__ Wp1,
                 const u32* __restrict__ Ap2, const u32* __restrict__ Wp2,
                 const float* __restrict__ bias1, const float* __restrict__ bias2,
                 float* __restrict__ C, int N) {
  __shared__ __align__(16) u32 bbuf[2][4 * 256];   // 2 x 4 tiles x 1KB

  const int lane = threadIdx.x & 31;
  const int wv   = threadIdx.x >> 5;   // 0..7 : M tile-pair (32 rows each)
  const int nb   = blockIdx.x;         // 64-wide N strip
  const int mtA  = wv << 1;
  const int KTtot = Ap2 ? (2 * KT32) : KT32;

  v8f acc[2][4];
#pragma unroll
  for (int i = 0; i < 2; ++i)
#pragma unroll
    for (int t = 0; t < 4; ++t) acc[i][t] = (v8f)(0.0f);

  const u32 aBase0 = ((u32)(mtA * KT32) * 32u + (u32)lane) * 8u;
  const u32 aBase1 = ((u32)((mtA + 1) * KT32) * 32u + (u32)lane) * 8u;

  // prologue: stage k = 0
  if (wv < 4) {
    stage_tile(Wp1, (nb << 2) + wv, 0, &bbuf[0][wv * 256], lane);
  }

  for (int k = 0; k < KTtot; ++k) {
    const int kk = k & (KT32 - 1);
    const u32* Ap = (k < KT32) ? Ap1 : Ap2;

    if (wv < 4) {
      asm volatile("s_wait_asynccnt 0x0" ::: "memory");   // stage(k) landed in LDS
    }
    __syncthreads();                                       // publish stage(k); prev reads done
    if (wv < 4 && (k + 1) < KTtot) {
      const u32* Wn = ((k + 1) < KT32) ? Wp1 : Wp2;
      const int  kn = (k + 1) & (KT32 - 1);
      stage_tile(Wn, (nb << 2) + wv, kn, &bbuf[(k + 1) & 1][wv * 256], lane);
    }

    v8u a0 = *(const v8u*)(Ap + aBase0 + (u32)kk * 256u);
    v8u a1 = *(const v8u*)(Ap + aBase1 + (u32)kk * 256u);
    v16bf av0 = __builtin_bit_cast(v16bf, a0);
    v16bf av1 = __builtin_bit_cast(v16bf, a1);
#pragma unroll
    for (int t = 0; t < 4; ++t) {
      v8u bw = *(const v8u*)&bbuf[k & 1][t * 256 + lane * 8];
      v16bf bv = __builtin_bit_cast(v16bf, bw);
      acc[0][t] = __builtin_amdgcn_wmma_f32_16x16x32_bf16(
          false, av0, false, bv, (short)0, acc[0][t], false, false);
      acc[1][t] = __builtin_amdgcn_wmma_f32_16x16x32_bf16(
          false, av1, false, bv, (short)0, acc[1][t], false, false);
    }
  }

  // C/D fragment: VGPR r -> M = r (lanes<16) or 8+r (lanes>=16); N = lane&15
  const int colLane = lane & 15;
  const int rowOff  = (lane >> 4) << 3;
#pragma unroll
  for (int t = 0; t < 4; ++t) {
    const int col = (nb << 6) + (t << 4) + colLane;
    float bv = 0.0f;
    if (bias1) bv += bias1[col];
    if (bias2) bv += bias2[col];
#pragma unroll
    for (int r = 0; r < 8; ++r) {
      const int row = (mtA << 4) + rowOff + r;
      C[(u32)row * (u32)N + (u32)col]        = acc[0][t][r] + bv;
      C[(u32)(row + 16) * (u32)N + (u32)col] = acc[1][t][r] + bv;
    }
  }
}

// ---------- SLSTM cell elementwise update ----------
__global__ void slstm_cell_kernel(const float* __restrict__ gates,
                                  const float* __restrict__ x_t,      // [B] or nullptr
                                  const float* __restrict__ Wih1,     // [4H] or nullptr
                                  float* __restrict__ syn, float* __restrict__ mem,
                                  const float* __restrict__ thr,
                                  u16* __restrict__ memPack,
                                  u16* __restrict__ spkPack,          // nullable
                                  float* __restrict__ spk_out) {      // nullable
  int idx = blockIdx.x * blockDim.x + threadIdx.x;    // < B*H
  int b = idx >> 10;
  int h = idx & 1023;
  const u32 gb = (u32)b * G4H;
  float gi = gates[gb + h];
  float gf = gates[gb + HID + h];
  float gg = gates[gb + 2 * HID + h];
  float go = gates[gb + 3 * HID + h];
  if (x_t) {  // layer 1: rank-1 input projection folded in
    float xv = x_t[b];
    gi += xv * Wih1[h];
    gf += xv * Wih1[HID + h];
    gg += xv * Wih1[2 * HID + h];
    go += xv * Wih1[3 * HID + h];
  }
  float th = thr[h];
  float m = mem[idx], s = syn[idx];
  float reset = (m - th) > 0.0f ? 1.0f : 0.0f;
  float si = sigmoidf_(gi), sf = sigmoidf_(gf), so = sigmoidf_(go);
  float sn = sf * s + si * tanhf(gg);
  float mn = so * tanhf(sn) - reset * th;
  syn[idx] = sn;
  mem[idx] = mn;
  float spk = (mn - th) > 0.0f ? 1.0f : 0.0f;

  u32 p = apack_idx(b, h);
  memPack[p] = f2bf(mn);
  if (spkPack) spkPack[p] = f2bf(spk);
  if (spk_out) spk_out[(u32)b * HID + h] = spk;
}

// ---------- Leaky #1: mo1 = clip(beta1)*mo1 + cur1 ----------
__global__ void leaky1_kernel(const float* __restrict__ cur1,
                              const float* __restrict__ beta1,
                              float* __restrict__ mo1) {
  int idx = blockIdx.x * blockDim.x + threadIdx.x;    // < B*H
  int h = idx & 1023;
  float b = beta1[h];
  b = fminf(fmaxf(b, 0.0f), 1.0f);
  mo1[idx] = b * mo1[idx] + cur1[idx];
}

// ---------- fc2 + Leaky #2 + mean: one wave per batch row ----------
__global__ void fc2_kernel(const float* __restrict__ mo1,
                           const float* __restrict__ fc2_W,   // [O,H]
                           const float* __restrict__ fc2_b,
                           const float* __restrict__ beta2,
                           float* __restrict__ mo2,           // [B,O] state
                           float* __restrict__ cur_out,       // [B,O] this step
                           float* __restrict__ mem_out) {     // [B]   this step
  int b = blockIdx.x;
  int lane = threadIdx.x;
  __shared__ float red[32];
  float mval = 0.0f;
  if (lane < OUTD) {
    const float* wr = fc2_W + (u32)lane * HID;
    const float* ar = mo1 + (u32)b * HID;
    float acc = 0.0f;
    for (int k = 0; k < HID; k += 4) {
      acc += ar[k] * wr[k] + ar[k + 1] * wr[k + 1] +
             ar[k + 2] * wr[k + 2] + ar[k + 3] * wr[k + 3];
    }
    acc += fc2_b[lane];
    float bb = fminf(fmaxf(beta2[lane], 0.0f), 1.0f);
    mval = bb * mo2[(u32)b * OUTD + lane] + acc;
    mo2[(u32)b * OUTD + lane] = mval;
    cur_out[(u32)b * OUTD + lane] = acc;
  }
  red[lane] = (lane < OUTD) ? mval : 0.0f;
  __syncthreads();
  if (lane == 0) {
    float s = 0.0f;
    for (int i = 0; i < OUTD; ++i) s += red[i];
    mem_out[b] = s * (1.0f / OUTD);
  }
}

// ---------- host ----------

extern "C" void kernel_launch(void* const* d_in, const int* in_sizes, int n_in,
                              void* d_out, int out_size, void* d_ws, size_t ws_size,
                              hipStream_t stream) {
  const float* x     = (const float*)d_in[0];
  const float* Wih1  = (const float*)d_in[1];
  const float* Whh1  = (const float*)d_in[2];
  const float* bih1  = (const float*)d_in[3];
  const float* bhh1  = (const float*)d_in[4];
  const float* thr1  = (const float*)d_in[5];
  const float* Wih2  = (const float*)d_in[6];
  const float* Whh2  = (const float*)d_in[7];
  const float* bih2  = (const float*)d_in[8];
  const float* bhh2  = (const float*)d_in[9];
  const float* thr2  = (const float*)d_in[10];
  const float* Wih3  = (const float*)d_in[11];
  const float* Whh3  = (const float*)d_in[12];
  const float* bih3  = (const float*)d_in[13];
  const float* bhh3  = (const float*)d_in[14];
  const float* thr3  = (const float*)d_in[15];
  const float* fc1_W = (const float*)d_in[16];
  const float* fc1_b = (const float*)d_in[17];
  const float* fc2_W = (const float*)d_in[18];
  const float* fc2_b = (const float*)d_in[19];
  const float* beta1 = (const float*)d_in[20];
  const float* beta2 = (const float*)d_in[22];

  float* cur_rec = (float*)d_out;                                   // [T,B,O]
  float* mem_rec = cur_rec + (size_t)T_STEPS * BATCH * OUTD;        // [T,B,1]
  float* spk_rec = mem_rec + (size_t)T_STEPS * BATCH;               // [T,B,H]

  char* ws = (char*)d_ws;
  size_t off = 0;
  auto alloc = [&](size_t bytes) -> void* {
    void* p = ws + off;
    off += (bytes + 255) & ~(size_t)255;
    return p;
  };

  const size_t WBIG = (size_t)G4H * HID * sizeof(u16);      // packed bf16 big weight
  u16* WhhP1 = (u16*)alloc(WBIG);
  u16* WihP2 = (u16*)alloc(WBIG);
  u16* WhhP2 = (u16*)alloc(WBIG);
  u16* WihP3 = (u16*)alloc(WBIG);
  u16* WhhP3 = (u16*)alloc(WBIG);
  u16* fc1P  = (u16*)alloc((size_t)HID * HID * sizeof(u16));
  float* gates = (float*)alloc((size_t)BATCH * G4H * sizeof(float));
  float* cur1  = (float*)alloc((size_t)BATCH * HID * sizeof(float));

  // zero-initialized region (states + packed activations), memset each launch
  size_t zeroStart = off;
  const size_t APACK = (size_t)BATCH * HID * sizeof(u16);
  u16* memP1 = (u16*)alloc(APACK);
  u16* memP2 = (u16*)alloc(APACK);
  u16* memP3 = (u16*)alloc(APACK);
  u16* spkP1 = (u16*)alloc(APACK);
  u16* spkP2 = (u16*)alloc(APACK);
  const size_t STATE = (size_t)BATCH * HID * sizeof(float);
  float* syn1 = (float*)alloc(STATE);
  float* mem1 = (float*)alloc(STATE);
  float* syn2 = (float*)alloc(STATE);
  float* mem2 = (float*)alloc(STATE);
  float* syn3 = (float*)alloc(STATE);
  float* mem3 = (float*)alloc(STATE);
  float* mo1  = (float*)alloc(STATE);
  float* mo2  = (float*)alloc((size_t)BATCH * OUTD * sizeof(float));
  size_t zeroEnd = off;
  (void)ws_size; (void)in_sizes; (void)n_in; (void)out_size;

  hipMemsetAsync(ws + zeroStart, 0, zeroEnd - zeroStart, stream);

  // Pack weights to WMMA B-fragment bf16 layout (L2-resident thereafter).
  {
    const int NKb = G4H * HID;
    dim3 blk(256), grd((NKb + 255) / 256);
    pack_w_kernel<<<grd, blk, 0, stream>>>(Whh1, WhhP1, NKb);
    pack_w_kernel<<<grd, blk, 0, stream>>>(Wih2, WihP2, NKb);
    pack_w_kernel<<<grd, blk, 0, stream>>>(Whh2, WhhP2, NKb);
    pack_w_kernel<<<grd, blk, 0, stream>>>(Wih3, WihP3, NKb);
    pack_w_kernel<<<grd, blk, 0, stream>>>(Whh3, WhhP3, NKb);
    const int NKf = HID * HID;
    pack_w_kernel<<<(NKf + 255) / 256, blk, 0, stream>>>(fc1_W, fc1P, NKf);
  }

  const dim3 cellGrid((BATCH * HID) / 256), cellBlk(256);
  const dim3 gemmBlk(256);
  const dim3 gemmGridBig(G4H >> 6);    // 64 blocks: one per 64-col strip, 8 waves cover M
  const dim3 gemmGridFc1(HID >> 6);    // 16 blocks

  for (int t = 0; t < T_STEPS; ++t) {
    const float* xt = x + (size_t)t * BATCH;

    // layer 1: gates = mem1 @ Whh1^T + bih1 + bhh1   (ih rank-1 term in cell)
    gemm_bf16_kernel<<<gemmGridBig, gemmBlk, 0, stream>>>(
        (const u32*)memP1, (const u32*)WhhP1, nullptr, nullptr, bih1, bhh1, gates, G4H);
    slstm_cell_kernel<<<cellGrid, cellBlk, 0, stream>>>(
        gates, xt, Wih1, syn1, mem1, thr1, memP1, spkP1, nullptr);

    // layer 2: gates = spk1 @ Wih2^T + mem2 @ Whh2^T + bih2 + bhh2
    gemm_bf16_kernel<<<gemmGridBig, gemmBlk, 0, stream>>>(
        (const u32*)spkP1, (const u32*)WihP2, (const u32*)memP2, (const u32*)WhhP2,
        bih2, bhh2, gates, G4H);
    slstm_cell_kernel<<<cellGrid, cellBlk, 0, stream>>>(
        gates, nullptr, nullptr, syn2, mem2, thr2, memP2, spkP2, nullptr);

    // layer 3
    gemm_bf16_kernel<<<gemmGridBig, gemmBlk, 0, stream>>>(
        (const u32*)spkP2, (const u32*)WihP3, (const u32*)memP3, (const u32*)WhhP3,
        bih3, bhh3, gates, G4H);
    slstm_cell_kernel<<<cellGrid, cellBlk, 0, stream>>>(
        gates, nullptr, nullptr, syn3, mem3, thr3, memP3, nullptr,
        spk_rec + (size_t)t * BATCH * HID);

    // fc1: cur1 = mem3 @ fc1_W^T + fc1_b
    gemm_bf16_kernel<<<gemmGridFc1, gemmBlk, 0, stream>>>(
        (const u32*)memP3, (const u32*)fc1P, nullptr, nullptr, fc1_b, nullptr, cur1, HID);
    leaky1_kernel<<<cellGrid, cellBlk, 0, stream>>>(cur1, beta1, mo1);

    // fc2 + leaky2 + mean
    fc2_kernel<<<BATCH, 32, 0, stream>>>(
        mo1, fc2_W, fc2_b, beta2, mo2,
        cur_rec + (size_t)t * BATCH * OUTD, mem_rec + (size_t)t * BATCH);
  }
}
